// SSIM_51436528337572
// MI455X (gfx1250) — matrix operand order repaired and checked by reading, not compiled, not run on hardware
//
#include <hip/hip_runtime.h>

typedef __attribute__((ext_vector_type(16))) _Float16 v16h;
typedef __attribute__((ext_vector_type(8)))  float    v8f;
typedef __attribute__((ext_vector_type(4)))  unsigned v4u;
typedef __attribute__((ext_vector_type(4)))  unsigned uv4;
typedef __attribute__((ext_vector_type(8)))  int      iv8;
typedef __attribute__((ext_vector_type(4)))  int      iv4;

#define TS 64     // output tile edge
#define IR 80     // padded input tile edge (64 + 2*5 halo, padded to 80)
#define HR 80     // padded intermediate rows (column-major [col][row])

#if defined(__has_builtin)
#if __has_builtin(__builtin_amdgcn_tensor_load_to_lds)
#define HAVE_TDM 1
#endif
#endif

__device__ inline unsigned pack2h(float lo, float hi) {
    union { _Float16 f; unsigned short s; } a, b;
    a.f = (_Float16)lo; b.f = (_Float16)hi;
    return (unsigned)a.s | ((unsigned)b.s << 16);
}

#ifdef HAVE_TDM
// 2D TDM load: copies tileH x tileW f32 elements (row stride = strideElems)
// starting at gsrc into LDS at byte offset ldsAddr, rows stored contiguously
// (tileW == LDS row stride). Reads beyond tensorW/tensorH return zero, which
// implements the convolution's zero padding at the right/bottom image edges.
__device__ inline void tdm_load_2d(unsigned ldsAddr, const float* gsrc,
                                   unsigned tensorW, unsigned tensorH,
                                   unsigned tileW, unsigned tileH,
                                   unsigned strideElems) {
    unsigned long long ga = (unsigned long long)(const void*)gsrc;
    uv4 g0;
    g0.x = 0x1u;                                   // count=1 valid, user D#
    g0.y = ldsAddr;                                // lds_addr (bytes)
    g0.z = (unsigned)(ga & 0xFFFFFFFFull);         // global_addr[31:0]
    g0.w = (unsigned)((ga >> 32) & 0x1FFFFFFull)   // global_addr[56:32]
         | (2u << 30);                             // type=2 ("image")
    iv8 g1;
    g1[0] = (int)(2u << 16);                       // data_size=2 -> 4 bytes
    g1[1] = (int)((tensorW & 0xFFFFu) << 16);      // tensor_dim0[15:0]
    g1[2] = (int)(((tensorW >> 16) & 0xFFFFu) |    // tensor_dim0[31:16]
                  ((tensorH & 0xFFFFu) << 16));    // tensor_dim1[15:0]
    g1[3] = (int)(((tensorH >> 16) & 0xFFFFu) |    // tensor_dim1[31:16]
                  ((tileW & 0xFFFFu) << 16));      // tile_dim0
    g1[4] = (int)(tileH & 0xFFFFu);                // tile_dim1 (tile_dim2=0)
    g1[5] = (int)strideElems;                      // tensor_dim0_stride[31:0]
    g1[6] = 0;                                     // stride hi / dim1_stride lo
    g1[7] = 0;
    iv4 g2 = {0, 0, 0, 0};                         // 2D: groups 2/3 unused
    iv4 g3 = {0, 0, 0, 0};
    iv8 g4 = {0, 0, 0, 0, 0, 0, 0, 0};             // unused trailing group
    __builtin_amdgcn_tensor_load_to_lds(g0, g1, g2, g3, g4, 0);
}
#endif

__global__ __launch_bounds__(256) void ssim_main(const float* __restrict__ img1,
                                                 const float* __restrict__ img2,
                                                 const float* __restrict__ win,
                                                 float* __restrict__ partial) {
    __shared__ __align__(16) float    sIn1[IR * IR];
    __shared__ __align__(16) float    sIn2[IR * IR];
    __shared__ __align__(16) _Float16 sH[5][TS * HR];
    __shared__ float sRed[8];

    const int tid   = threadIdx.x;
    const int lane  = tid & 31;
    const int wave  = tid >> 5;
    const int lhalf = lane >> 4;   // 0: lanes 0-15, 1: lanes 16-31
    const int lm    = lane & 15;

    const int plane = blockIdx.x >> 6;      // 48 planes (16 batch * 3 chan)
    const int tIdx  = blockIdx.x & 63;      // 8x8 tiles of 64x64
    const int ty    = tIdx >> 3;
    const int tx    = tIdx & 7;
    const int y0    = ty * TS;
    const int x0    = tx * TS;
    const long planeOff = (long)plane * 512 * 512;

#ifdef HAVE_TDM
    const bool interior = (tx > 0) && (ty > 0);   // halo start is in-bounds
#else
    const bool interior = false;
#endif

    // --- Issue the tile DMAs FIRST so the global->LDS transfer overlaps the
    // filter-fragment initialization below. Wave 0 moves img1, wave 1 img2.
#ifdef HAVE_TDM
    if (interior && wave < 2) {
        const float* src = (wave == 0) ? img1 : img2;
        float* dstLds    = (wave == 0) ? sIn1 : sIn2;
        unsigned ldsAddr = (unsigned)(unsigned long long)(void*)dstLds;
        const float* gsrc = src + planeOff + (long)(y0 - 5) * 512 + (x0 - 5);
        unsigned remW = (unsigned)(512 - (x0 - 5));   // OOB -> zeros
        unsigned remH = (unsigned)(512 - (y0 - 5));
        tdm_load_2d(ldsAddr, gsrc, remW, remH, IR, IR, 512u);
    }
#endif

    // --- 1D gaussian from separable 2D window: g1d[j] = w2d[5][j]/sqrt(w2d[5][5])
    float g[11];
    {
        float inv = rsqrtf(win[5 * 11 + 5]);
        #pragma unroll
        for (int j = 0; j < 11; ++j) g[j] = win[5 * 11 + j] * inv;
    }

    // --- Filter fragments in WMMA register layouts (banded matrices).
    // Bh: 32x16 B operand, Bh[k][n] = g[k-n]   (horizontal pass, data in A)
    // Av: 16x32 A operand, Av[m][k] = g[k-m]   (vertical pass, data in B)
    v16h Bh, Av;
    #pragma unroll
    for (int v = 0; v < 8; ++v) {
        #pragma unroll
        for (int p = 0; p < 2; ++p) {
            int kb = lhalf * 8 + (v & 3) * 2 + (v >> 2) * 16 + p;
            int db = kb - lm;
            float fb = 0.f;
            #pragma unroll
            for (int t = 0; t < 11; ++t) if (db == t) fb = g[t];
            Bh[2 * v + p] = (_Float16)fb;

            int ka = lhalf * 16 + 2 * v + p;
            int da = ka - lm;
            float fa = 0.f;
            #pragma unroll
            for (int t = 0; t < 11; ++t) if (da == t) fa = g[t];
            Av[2 * v + p] = (_Float16)fa;
        }
    }

    // --- Complete staging.
    if (interior) {
#ifdef HAVE_TDM
        if (wave < 2) __builtin_amdgcn_s_wait_tensorcnt(0);
#endif
    } else {
        // Edge tiles: manual bounds-checked staging (zero halo).
        for (int i = tid; i < IR * IR; i += 256) {
            int r = i / IR, cc = i - r * IR;
            int gr = y0 - 5 + r, gc = x0 - 5 + cc;
            float a = 0.f, b = 0.f;
            if ((unsigned)gr < 512u && (unsigned)gc < 512u) {
                long o = planeOff + (long)gr * 512 + gc;
                a = img1[o]; b = img2[o];
            }
            sIn1[i] = a;
            sIn2[i] = b;
        }
    }
    __syncthreads();

    // --- Horizontal pass: 5 row-bands x 4 col-groups, per-wave uniform work.
    for (int item = wave; item < 20; item += 8) {
        int r = item >> 2, c = item & 3;
        int row = r * 16 + lm, colBase = c * 16;
        union { v16h h; unsigned u[8]; } a1, a2;
        #pragma unroll
        for (int v = 0; v < 8; ++v) {
            int k = lhalf * 8 + (v & 3) * 2 + (v >> 2) * 16;
            int idx = row * IR + colBase + k;          // 2 consecutive f32
            float2 d1 = *(const float2*)&sIn1[idx];
            float2 d2 = *(const float2*)&sIn2[idx];
            a1.u[v] = pack2h(d1.x, d1.y);              // v_cvt_pk_f16_f32
            a2.u[v] = pack2h(d2.x, d2.y);
        }
        v16h A1 = a1.h, A2 = a2.h;
        v16h Aq[5] = { A1, A2, A1 * A1, A2 * A2, A1 * A2 };   // packed f16 muls
        int hidx = (colBase + lm) * HR + r * 16 + lhalf * 8;  // column-major
        #pragma unroll
        for (int q = 0; q < 5; ++q) {
            v8f acc = {};
            acc = __builtin_amdgcn_wmma_f32_16x16x32_f16(
                false, Aq[q], false, Bh, (short)0, acc, false, false);
            v4u st;
            st.x = pack2h(acc[0], acc[1]);
            st.y = pack2h(acc[2], acc[3]);
            st.z = pack2h(acc[4], acc[5]);
            st.w = pack2h(acc[6], acc[7]);
            *(v4u*)&sH[q][hidx] = st;                 // ds_store_b128
        }
    }
    __syncthreads();

    // --- Vertical pass + fused SSIM map: 4 row-blocks x 4 col-groups.
    float lsum = 0.f;
    for (int item = wave; item < 16; item += 8) {
        int yb = item >> 2, c = item & 3;
        int col = c * 16 + lm;
        v8f res[5];
        #pragma unroll
        for (int q = 0; q < 5; ++q) {
            union { v16h h; unsigned u[8]; } bf;
            #pragma unroll
            for (int v = 0; v < 8; ++v) {
                int k = lhalf * 16 + 2 * v;            // 2 consecutive rows
                bf.u[v] = *(const unsigned*)&sH[q][col * HR + yb * 16 + k];
            }
            v8f acc = {};
            res[q] = __builtin_amdgcn_wmma_f32_16x16x32_f16(
                false, Av, false, bf.h, (short)0, acc, false, false);
        }
        const float C1 = 1.0e-4f, C2 = 9.0e-4f;
        #pragma unroll
        for (int i = 0; i < 8; ++i) {
            float mu1 = res[0][i], mu2 = res[1][i];
            float mu1sq = mu1 * mu1, mu2sq = mu2 * mu2, mu12 = mu1 * mu2;
            float s1  = res[2][i] - mu1sq;
            float s2  = res[3][i] - mu2sq;
            float s12 = res[4][i] - mu12;
            float num = (2.f * mu12 + C1) * (2.f * s12 + C2);
            float den = (mu1sq + mu2sq + C1) * (s1 + s2 + C2);
            lsum += num / den;
        }
    }

    // --- Block reduction -> per-block partial
    #pragma unroll
    for (int off = 16; off > 0; off >>= 1) lsum += __shfl_down(lsum, off, 32);
    if (lane == 0) sRed[wave] = lsum;
    __syncthreads();
    if (tid == 0) {
        float s = 0.f;
        #pragma unroll
        for (int i = 0; i < 8; ++i) s += sRed[i];
        partial[blockIdx.x] = s;
    }
}

__global__ __launch_bounds__(256) void ssim_reduce(const float* __restrict__ partial,
                                                   float* __restrict__ out,
                                                   int n, float scale) {
    __shared__ float w[8];
    float t = 0.f;
    for (int i = threadIdx.x; i < n; i += 256) t += partial[i];
    #pragma unroll
    for (int off = 16; off > 0; off >>= 1) t += __shfl_down(t, off, 32);
    if ((threadIdx.x & 31) == 0) w[threadIdx.x >> 5] = t;
    __syncthreads();
    if (threadIdx.x == 0) {
        float r = 0.f;
        #pragma unroll
        for (int i = 0; i < 8; ++i) r += w[i];
        out[0] = r * scale;
    }
}

extern "C" void kernel_launch(void* const* d_in, const int* in_sizes, int n_in,
                              void* d_out, int out_size, void* d_ws, size_t ws_size,
                              hipStream_t stream) {
    const float* img1 = (const float*)d_in[0];   // [16][3][512][512] f32
    const float* img2 = (const float*)d_in[1];   // [16][3][512][512] f32
    const float* win  = (const float*)d_in[2];   // [3][1][11][11]   f32
    float* out     = (float*)d_out;              // single f32 (mean)
    float* partial = (float*)d_ws;               // 3072 f32 partials

    const int nBlocks = 16 * 3 * (512 / TS) * (512 / TS);   // 3072
    ssim_main<<<nBlocks, 256, 0, stream>>>(img1, img2, win, partial);
    ssim_reduce<<<1, 256, 0, stream>>>(partial, out, nBlocks,
                                       1.0f / (16.f * 3.f * 512.f * 512.f));
}
